// DKVMN_28638841930444
// MI455X (gfx1250) — compile-verified, matrix operand order checked by reading, not run on hardware
//
#include <hip/hip_runtime.h>
#include <cmath>

// ---------------------------------------------------------------------------
// DKVMN scan for MI455X (gfx1250): persistent wave32 kernel, all GEMMs on
// v_wmma_f32_16x16x32_bf16, m-state kept L2-resident, fused read+update,
// hardware v_tanh_f32 gates, async global->LDS double-buffered q stream,
// NT streaming loads/stores, coalesced staged output.
// ---------------------------------------------------------------------------

typedef __attribute__((ext_vector_type(16))) __bf16 v16bf;
typedef __attribute__((ext_vector_type(8)))  float  v8f;
typedef __attribute__((ext_vector_type(4)))  float  f32x4;
typedef __attribute__((ext_vector_type(4)))  int    i32x4;

#define T_STEPS 512
#define B_TOT   256
#define M_SLOTS 64
#define DK_DIM  64
#define DV_DIM  256
#define H_DIM   128
#define CAT_DIM 320           // DK + DV
#define BT      16            // batch tile = WMMA M
#define NWAVES  16
#define FS_ELEMS ((size_t)T_STEPS * (size_t)B_TOT * (size_t)H_DIM)

// Workspace layout (bf16 elements): pre-packed B-operand fragments.
// fragment = 512 bf16 (32 lanes x 16 halves), indexed [nt][kc].
#define KEY_OFF 0                      // key^T  : N=64  (4 nt),  K=64  (2 kc)
#define KEY_SZ  (4 * 2 * 512)
#define WE_OFF  (KEY_OFF + KEY_SZ)     // W_E^T  : N=256 (16 nt), K=256 (8 kc)
#define WE_SZ   (16 * 8 * 512)
#define WD_OFF  (WE_OFF + WE_SZ)       // W_D^T
#define WD_SZ   (16 * 8 * 512)
#define W1_OFF  (WD_OFF + WD_SZ)       // W1^T   : N=128 (8 nt),  K=320 (10 kc)
#define W1_SZ   (8 * 10 * 512)
#define WS_TOTAL (W1_OFF + W1_SZ)      // 176128 bf16 = 344 KB

// --- async global->LDS support (gfx1250 ASYNCcnt path) ----------------------
#if defined(__HIP_DEVICE_COMPILE__)
#if __has_builtin(__builtin_amdgcn_global_load_async_to_lds_b128) && \
    __has_builtin(__builtin_amdgcn_s_wait_asynccnt)
#define HAVE_ASYNC 1
#else
#define HAVE_ASYNC 0
#warning "gfx1250 async-to-LDS builtins unavailable; using synchronous q path"
#endif
#else
#define HAVE_ASYNC 0
#endif

#if HAVE_ASYNC
// prototype (from clang diagnostic): (int4 AS1* gaddr, int4 AS3* lds, i32 off, i32 cpol)
typedef __attribute__((address_space(1))) i32x4 g_i32x4;
typedef __attribute__((address_space(3))) i32x4 l_i32x4;
// copy 32 consecutive bytes (8 f32) global -> LDS via two async b128 ops;
// ISA 08: IOFFSET is added to both the global and the LDS address.
__device__ inline void async_copy_8f(const float* gsrc, float* ldst) {
    __builtin_amdgcn_global_load_async_to_lds_b128((g_i32x4*)gsrc, (l_i32x4*)ldst, 0, 0);
    __builtin_amdgcn_global_load_async_to_lds_b128((g_i32x4*)gsrc, (l_i32x4*)ldst, 16, 0);
}
#endif

// --- fast transcendentals ---------------------------------------------------
__device__ inline float tanh_f(float x) {
#if __has_builtin(__builtin_amdgcn_tanhf)
    return __builtin_amdgcn_tanhf(x);            // v_tanh_f32 (gfx1250 trans op)
#elif __has_builtin(__builtin_amdgcn_tanh_f32)
    return __builtin_amdgcn_tanh_f32(x);
#else
    float e = __expf(2.0f * x);                  // tanh = 1 - 2/(e^{2x}+1)
    return 1.0f - 2.0f * __builtin_amdgcn_rcpf(e + 1.0f);
#endif
}
__device__ inline float sigmoid_f(float x) {     // exact identity, 1 trans op
    return 0.5f * tanh_f(0.5f * x) + 0.5f;
}

// --- ISA 7.12.2 operand layouts (wave32) -----------------------------------
// A (16x32 bf16): lane L holds row M = L&15; dword d holds K pair:
//   K = 2*(d&3) + 16*(d>>2) + 8*(L>>4)
// B (32x16 bf16): lane L holds col N = L&15; half i holds:
//   K = (i&7) + 16*(i>>3) + 8*(L>>4)
__device__ __host__ inline int bfrag_k(int lane, int i) {
    return (i & 7) + ((i >> 3) << 4) + ((lane >> 4) << 3);
}

__device__ inline v16bf load_afrag(const __bf16* base, int stride, int kc, int lane) {
    union { v16bf v; unsigned u[8]; } r;
    const __bf16* p = base + (lane & 15) * stride + kc * 32;
    const int khi = (lane >> 4) << 3;
#pragma unroll
    for (int d = 0; d < 8; ++d) {
        int k = ((d & 3) << 1) + ((d >> 2) << 4) + khi;
        r.u[d] = *(const unsigned*)(p + k);      // bf16 pair (K, K+1) -> b128 merged
    }
    return r.v;
}

__device__ inline v16bf load_bfrag(const __bf16* frag, int lane) {
    union { v16bf v; uint4 u[2]; } r;
    const uint4* p = (const uint4*)(frag + lane * 16);   // 32B per lane
    r.u[0] = p[0];
    r.u[1] = p[1];
    return r.v;
}

// ---------------------------------------------------------------------------
// Kernel 1: pack weights (f32 row-major, out-features x in-features) into
// bf16 WMMA B-fragments.  One thread per output bf16 element.
// ---------------------------------------------------------------------------
__global__ void dkvmn_pack_weights(const float* __restrict__ key,
                                   const float* __restrict__ W_E,
                                   const float* __restrict__ W_D,
                                   const float* __restrict__ W1,
                                   __bf16* __restrict__ ws) {
    int idx = blockIdx.x * blockDim.x + threadIdx.x;
    if (idx >= WS_TOTAL) return;
    const float* src;
    int Kin, base;
    if (idx < WE_OFF)      { src = key; Kin = 64;  base = KEY_OFF; }
    else if (idx < WD_OFF) { src = W_E; Kin = 256; base = WE_OFF; }
    else if (idx < W1_OFF) { src = W_D; Kin = 256; base = WD_OFF; }
    else                   { src = W1;  Kin = 320; base = W1_OFF; }
    int local  = idx - base;
    int lane   = (local >> 4) & 31;
    int i      = local & 15;
    int fragid = local >> 9;             // nt * (Kin/32) + kc
    int KC     = Kin >> 5;
    int nt     = fragid / KC;
    int kc     = fragid - nt * KC;
    int n      = nt * 16 + (lane & 15);
    int k      = kc * 32 + bfrag_k(lane, i);
    ws[idx] = (__bf16)src[n * Kin + k];  // B[k][n] = W[n][k]  (W^T GEMM)
}

// ---------------------------------------------------------------------------
// Kernel 2: persistent scan.  grid = 16 WGs (one 16-batch tile each),
// block = 512 threads = 16 wave32 waves.
// ---------------------------------------------------------------------------
__global__ __launch_bounds__(512) void dkvmn_scan(
    const float* __restrict__ q,     // (T, B, DV)
    const float* __restrict__ m0,    // (B, M, DV)  -- read only
    const float* __restrict__ bE,    // (DV)
    const float* __restrict__ bD,    // (DV)
    const float* __restrict__ b1,    // (H)
    const __bf16* __restrict__ ws,   // packed weight fragments
    float* __restrict__ out) {       // fs (T*B, H) then m_final (B, M, DV)

    __shared__ __bf16 sQ[BT][DV_DIM];     // q_t tile, bf16 row-major
    __shared__ __bf16 sCat[BT][CAT_DIM];  // [r, k_t] bf16
    __shared__ float  sW[BT][M_SLOTS];    // logits -> softmax weights
    __shared__ float  sE[BT][DV_DIM];     // sigmoid gate
    __shared__ float  sA[BT][DV_DIM];     // tanh write vector
    __shared__ float  sF[BT][H_DIM];      // staged f-tile for coalesced store
#if HAVE_ASYNC
    __shared__ float  sQstage[2][BT * DV_DIM];  // double-buffered async q stage
#endif

    const int tid  = threadIdx.x;
    const int lane = tid & 31;
    const int wave = tid >> 5;
    const int bg0  = blockIdx.x * BT;

    float* fs   = out;
    float* mBlk = out + FS_ELEMS + (size_t)bg0 * M_SLOTS * DV_DIM;

#if HAVE_ASYNC
    // kick off t=0 q tile while we copy the m state below
    async_copy_8f(q + (size_t)bg0 * DV_DIM + tid * 8, &sQstage[0][tid * 8]);
#endif

    // --- init m state in the m_final output region (never mutate d_in) ----
    {
        const f32x4* src = (const f32x4*)(m0 + (size_t)bg0 * M_SLOTS * DV_DIM);
        f32x4* dst = (f32x4*)mBlk;
        const int n4 = BT * M_SLOTS * DV_DIM / 4;      // 65536 f32x4
        for (int i = tid; i < n4; i += 512) dst[i] = src[i];
    }
    __syncthreads();

    const float biasE = bE[wave * 16 + (lane & 15)];
    const float biasD = bD[wave * 16 + (lane & 15)];
    const float bias1 = (wave < 8) ? b1[wave * 16 + (lane & 15)] : 0.0f;

    for (int t = 0; t < T_STEPS; ++t) {
        // ---- phase 1: materialize q_t tile as bf16 in LDS ----------------
        {
            const int flat = tid * 8;                 // 8 consecutive f32/thread
            const int row = flat >> 8, col = flat & 255;
            f32x4 v0, v1;
#if HAVE_ASYNC
            // this thread async-loaded exactly this slice last step; its own
            // ASYNCcnt==0 makes the LDS writes visible to itself -> no barrier.
            __builtin_amdgcn_s_wait_asynccnt(0);
            const f32x4* sp = (const f32x4*)&sQstage[t & 1][flat];
            v0 = sp[0]; v1 = sp[1];
            if (t + 1 < T_STEPS)                      // stream next tile now
                async_copy_8f(q + ((size_t)(t + 1) * B_TOT + bg0) * DV_DIM + flat,
                              &sQstage[(t + 1) & 1][flat]);
#else
            const f32x4* qs = (const f32x4*)(q + ((size_t)t * B_TOT + bg0) * DV_DIM + flat);
            v0 = __builtin_nontemporal_load(&qs[0]);
            v1 = __builtin_nontemporal_load(&qs[1]);
            if (t + 1 < T_STEPS) {
                const char* np = (const char*)(q + ((size_t)(t + 1) * B_TOT + bg0) * DV_DIM)
                                 + (size_t)tid * 32;
                __builtin_prefetch(np, 0, 1);         // global_prefetch_b8
            }
#endif
            sQ[row][col + 0] = (__bf16)v0.x;
            sQ[row][col + 1] = (__bf16)v0.y;
            sQ[row][col + 2] = (__bf16)v0.z;
            sQ[row][col + 3] = (__bf16)v0.w;
            sQ[row][col + 4] = (__bf16)v1.x;
            sQ[row][col + 5] = (__bf16)v1.y;
            sQ[row][col + 6] = (__bf16)v1.z;
            sQ[row][col + 7] = (__bf16)v1.w;
            if (col < DK_DIM) {                       // k_t tail of cat
                sCat[row][DV_DIM + col + 0] = (__bf16)v0.x;
                sCat[row][DV_DIM + col + 1] = (__bf16)v0.y;
                sCat[row][DV_DIM + col + 2] = (__bf16)v0.z;
                sCat[row][DV_DIM + col + 3] = (__bf16)v0.w;
                sCat[row][DV_DIM + col + 4] = (__bf16)v1.x;
                sCat[row][DV_DIM + col + 5] = (__bf16)v1.y;
                sCat[row][DV_DIM + col + 6] = (__bf16)v1.z;
                sCat[row][DV_DIM + col + 7] = (__bf16)v1.w;
            }
        }
        __syncthreads();

        // ---- phase 2: WMMA gates e,a (all waves) + logits (waves 0-3) ----
        {
            v8f accE = {}; v8f accA = {};
            const __bf16* fE = ws + WE_OFF + (size_t)wave * 8 * 512;
            const __bf16* fD = ws + WD_OFF + (size_t)wave * 8 * 512;
#pragma unroll
            for (int kc = 0; kc < 8; ++kc) {
                v16bf a  = load_afrag(&sQ[0][0], DV_DIM, kc, lane);
                v16bf be = load_bfrag(fE + kc * 512, lane);
                v16bf bd = load_bfrag(fD + kc * 512, lane);
                accE = __builtin_amdgcn_wmma_f32_16x16x32_bf16(
                           false, a, false, be, (short)0, accE, false, false);
                accA = __builtin_amdgcn_wmma_f32_16x16x32_bf16(
                           false, a, false, bd, (short)0, accA, false, false);
            }
            const int row0 = (lane >> 4) << 3;
            const int col  = wave * 16 + (lane & 15);
#pragma unroll
            for (int j = 0; j < 8; ++j) {
                sE[row0 + j][col] = sigmoid_f(accE[j] + biasE);
                sA[row0 + j][col] = tanh_f(accA[j] + biasD);
            }
            if (wave < 4) {                            // uniform per wave: EXEC all-1
                v8f accL = {};
                const __bf16* fK = ws + KEY_OFF + (size_t)wave * 2 * 512;
#pragma unroll
                for (int kc = 0; kc < 2; ++kc) {
                    v16bf a  = load_afrag(&sQ[0][0], DV_DIM, kc, lane);
                    v16bf bk = load_bfrag(fK + kc * 512, lane);
                    accL = __builtin_amdgcn_wmma_f32_16x16x32_bf16(
                               false, a, false, bk, (short)0, accL, false, false);
                }
#pragma unroll
                for (int j = 0; j < 8; ++j) sW[row0 + j][col] = accL[j];
            }
        }
        __syncthreads();

        // ---- phase 3: softmax over M=64 slots, one WAVE per batch row ----
        {
            const int b = wave;
            float l0 = sW[b][lane];
            float l1 = sW[b][lane + 32];
            float mx = fmaxf(l0, l1);
#pragma unroll
            for (int off = 16; off > 0; off >>= 1)
                mx = fmaxf(mx, __shfl_xor(mx, off, 32));
            float e0v = __expf(l0 - mx);
            float e1v = __expf(l1 - mx);
            float sum = e0v + e1v;
#pragma unroll
            for (int off = 16; off > 0; off >>= 1)
                sum += __shfl_xor(sum, off, 32);
            float inv = __builtin_amdgcn_rcpf(sum);
            sW[b][lane]      = e0v * inv;
            sW[b][lane + 32] = e1v * inv;
        }
        __syncthreads();

        // ---- phase 4: fused read r = w@m  +  m update (wave == batch) ----
        {
            const int b = wave;
            float* mrow = mBlk + (size_t)b * M_SLOTS * DV_DIM;
            const int v0 = lane * 4, v1 = 128 + lane * 4;
            f32x4 e0 = *(const f32x4*)&sE[b][v0];
            f32x4 e1 = *(const f32x4*)&sE[b][v1];
            f32x4 a0 = *(const f32x4*)&sA[b][v0];
            f32x4 a1 = *(const f32x4*)&sA[b][v1];
            f32x4 r0 = {0, 0, 0, 0}, r1 = {0, 0, 0, 0};
#pragma unroll 8
            for (int mI = 0; mI < M_SLOTS; ++mI) {
                float wv = sW[b][mI];                  // LDS broadcast
                f32x4* p0 = (f32x4*)(mrow + mI * DV_DIM + v0);
                f32x4* p1 = (f32x4*)(mrow + mI * DV_DIM + v1);
                f32x4 x0 = *p0, x1 = *p1;
                r0 += wv * x0;
                r1 += wv * x1;
                x0 = x0 * (1.0f - e0 * wv) + a0 * wv;
                x1 = x1 * (1.0f - e1 * wv) + a1 * wv;
                *p0 = x0; *p1 = x1;
            }
            sCat[b][v0 + 0] = (__bf16)r0.x; sCat[b][v0 + 1] = (__bf16)r0.y;
            sCat[b][v0 + 2] = (__bf16)r0.z; sCat[b][v0 + 3] = (__bf16)r0.w;
            sCat[b][v1 + 0] = (__bf16)r1.x; sCat[b][v1 + 1] = (__bf16)r1.y;
            sCat[b][v1 + 2] = (__bf16)r1.z; sCat[b][v1 + 3] = (__bf16)r1.w;
        }
        __syncthreads();

        // ---- phase 5: f = tanh([r,k_t] @ W1^T + b1), waves 0-7, stage ----
        if (wave < 8) {
            v8f acc = {};
            const __bf16* f1 = ws + W1_OFF + (size_t)wave * 10 * 512;
#pragma unroll
            for (int kc = 0; kc < 10; ++kc) {
                v16bf a  = load_afrag(&sCat[0][0], CAT_DIM, kc, lane);
                v16bf bw = load_bfrag(f1 + kc * 512, lane);
                acc = __builtin_amdgcn_wmma_f32_16x16x32_bf16(
                          false, a, false, bw, (short)0, acc, false, false);
            }
            const int row0 = (lane >> 4) << 3;
            const int col  = wave * 16 + (lane & 15);
#pragma unroll
            for (int j = 0; j < 8; ++j)
                sF[row0 + j][col] = tanh_f(acc[j] + bias1);
        }
        __syncthreads();

        // ---- phase 6: coalesced NT store of the 16x128 f-tile ------------
        {
            int row = tid >> 5;
            int c4  = (tid & 31) << 2;
            f32x4 v = *(const f32x4*)&sF[row][c4];
            f32x4* dst = (f32x4*)(fs + ((size_t)t * B_TOT + bg0 + row) * H_DIM + c4);
            __builtin_nontemporal_store(v, dst);      // write-once stream -> NT
        }
        // no trailing barrier needed: next phase-1 writes sQ/sCat/its own
        // stage slice, whose last readers completed before the barrier above.
    }
}

// ---------------------------------------------------------------------------
extern "C" void kernel_launch(void* const* d_in, const int* in_sizes, int n_in,
                              void* d_out, int out_size, void* d_ws, size_t ws_size,
                              hipStream_t stream) {
    (void)in_sizes; (void)n_in; (void)out_size; (void)ws_size;
    const float* q    = (const float*)d_in[0];
    const float* m_kt = (const float*)d_in[1];
    const float* key  = (const float*)d_in[2];
    const float* W_E  = (const float*)d_in[3];
    const float* b_E  = (const float*)d_in[4];
    const float* W_D  = (const float*)d_in[5];
    const float* b_D  = (const float*)d_in[6];
    const float* W1   = (const float*)d_in[7];
    const float* b1   = (const float*)d_in[8];
    float*  out = (float*)d_out;
    __bf16* ws  = (__bf16*)d_ws;

    dkvmn_pack_weights<<<(WS_TOTAL + 255) / 256, 256, 0, stream>>>(
        key, W_E, W_D, W1, ws);
    dkvmn_scan<<<B_TOT / BT, NWAVES * 32, 0, stream>>>(
        q, m_kt, b_E, b_D, b1, ws, out);
}
